// AMK_Block_19902878450347
// MI455X (gfx1250) — matrix operand order repaired and checked by reading, not compiled, not run on hardware
//
#include <hip/hip_runtime.h>
#include <math.h>

typedef __attribute__((ext_vector_type(16))) __bf16 v16bf;
typedef __attribute__((ext_vector_type(8)))  __bf16 v8bf;
typedef __attribute__((ext_vector_type(8)))  float  v8f;

// ---------------- WMMA fragment loaders (LDS -> VGPR) ----------------
// A operand 16x32 bf16: lanes 0-15 = row M, elems 0..7 = K kh*8.., elems 8..15 = K 16+kh*8..
__device__ __forceinline__ v16bf frag_a(const __bf16* buf, int stride, int row0, int k0, int lane) {
  int lr = lane & 15;
  int kh = (lane >> 4) * 8;
  const __bf16* p = buf + (size_t)(row0 + lr) * stride + k0 + kh;
  v8bf lo = *(const v8bf*)p;
  v8bf hi = *(const v8bf*)(p + 16);
  v16bf r;
#pragma unroll
  for (int i = 0; i < 8; ++i) { r[i] = lo[i]; r[i + 8] = hi[i]; }
  return r;
}
// B operand 32x16 bf16 (stored as Bt[row=N][k]): lanes 0-15 = col N, elems 0..7 = K kh*8.., elems 8..15 = K 16+kh*8..
__device__ __forceinline__ v16bf frag_b(const __bf16* buf, int stride, int row0, int k0, int lane) {
  int lr = lane & 15;
  int kh = (lane >> 4) * 8;
  const __bf16* p = buf + (size_t)(row0 + lr) * stride + k0 + kh;
  v8bf lo = *(const v8bf*)p;
  v8bf hi = *(const v8bf*)(p + 16);
  v16bf r;
#pragma unroll
  for (int i = 0; i < 8; ++i) { r[i] = lo[i]; r[i + 8] = hi[i]; }
  return r;
}

#define WMMA_BF16(a, b, c) \
  __builtin_amdgcn_wmma_f32_16x16x32_bf16(false, (a), false, (b), (short)0, (c), false, false)

// ---------------- Generic bf16 GEMM: C[M,N] = A[M,K] * Bt[N,K]^T ----------------
// 256 threads = 8 waves. Block tile 128x256, K-step 64. Waves: 2(M) x 4(N), 64x64 each
// (16 WMMA per 8 fragment loads). Next K-tile is register-staged during compute.
__global__ __launch_bounds__(256) void AMK_gemm_bf16(const __bf16* __restrict__ A,
                                                     const __bf16* __restrict__ Bt,
                                                     float* __restrict__ C,
                                                     int M, int N, int K) {
  __shared__ __bf16 As[128][72];
  __shared__ __bf16 Bs[256][72];
  const int bm = blockIdx.y * 128, bn = blockIdx.x * 256;
  const int t = threadIdx.x, lane = t & 31, wave = t >> 5;
  const int wm = (wave & 1) * 64;
  const int wn = (wave >> 1) * 64;

  v8f acc[4][4];
#pragma unroll
  for (int i = 0; i < 4; ++i)
#pragma unroll
    for (int j = 0; j < 4; ++j) acc[i][j] = (v8f){0.f,0.f,0.f,0.f,0.f,0.f,0.f,0.f};

  v8bf ra[4], rb[8];

  auto gload = [&](int k0) {
#pragma unroll
    for (int i = 0; i < 4; ++i) {
      int idx = t + i * 256, row = idx >> 3, col = (idx & 7) * 8;
      ra[i] = *(const v8bf*)(A + (size_t)(bm + row) * K + k0 + col);
      if (k0 + 64 < K) __builtin_prefetch(A + (size_t)(bm + row) * K + k0 + 64 + col, 0, 1);
    }
#pragma unroll
    for (int i = 0; i < 8; ++i) {
      int idx = t + i * 256, row = idx >> 3, col = (idx & 7) * 8;
      rb[i] = *(const v8bf*)(Bt + (size_t)(bn + row) * K + k0 + col);
      if (k0 + 64 < K) __builtin_prefetch(Bt + (size_t)(bn + row) * K + k0 + 64 + col, 0, 1);
    }
  };
  auto lstore = [&]() {
#pragma unroll
    for (int i = 0; i < 4; ++i) {
      int idx = t + i * 256, row = idx >> 3, col = (idx & 7) * 8;
      *(v8bf*)&As[row][col] = ra[i];
    }
#pragma unroll
    for (int i = 0; i < 8; ++i) {
      int idx = t + i * 256, row = idx >> 3, col = (idx & 7) * 8;
      *(v8bf*)&Bs[row][col] = rb[i];
    }
  };
  auto compute = [&]() {
#pragma unroll
    for (int kk = 0; kk < 64; kk += 32) {
      v16bf af[4], bfr[4];
#pragma unroll
      for (int mi = 0; mi < 4; ++mi) af[mi]  = frag_a(&As[0][0], 72, wm + mi * 16, kk, lane);
#pragma unroll
      for (int ni = 0; ni < 4; ++ni) bfr[ni] = frag_b(&Bs[0][0], 72, wn + ni * 16, kk, lane);
#pragma unroll
      for (int mi = 0; mi < 4; ++mi)
#pragma unroll
        for (int ni = 0; ni < 4; ++ni) acc[mi][ni] = WMMA_BF16(af[mi], bfr[ni], acc[mi][ni]);
    }
  };

  gload(0);
  lstore();
  __syncthreads();
  for (int k0 = 0; k0 < K; k0 += 64) {
    bool has_next = (k0 + 64) < K;
    if (has_next) gload(k0 + 64);   // issue global loads; consumed after compute
    compute();
    if (has_next) {
      __syncthreads();
      lstore();
      __syncthreads();
    }
  }

  const int laneN = lane & 15, mOff = (lane >> 4) * 8;
#pragma unroll
  for (int mi = 0; mi < 4; ++mi)
#pragma unroll
    for (int ni = 0; ni < 4; ++ni)
#pragma unroll
      for (int r = 0; r < 8; ++r)
        C[(size_t)(bm + wm + mi * 16 + mOff + r) * N + bn + wn + ni * 16 + laneN] = acc[mi][ni][r];
}

// ---------------- RoPE + per-head RMS, build Qn/Kn [H][N][64] bf16 and Vt [H][64][N] bf16 ----
__global__ __launch_bounds__(256) void AMK_rope_norm(const float* __restrict__ qkv,
                                                     const float* __restrict__ cosb,
                                                     const float* __restrict__ sinb,
                                                     __bf16* __restrict__ Qn,
                                                     __bf16* __restrict__ Kn,
                                                     __bf16* __restrict__ Vt) {
  int id = blockIdx.x * 256 + threadIdx.x;   // 4096*16
  int n = id >> 4, h = id & 15;
  const float* q = qkv + ((size_t)n * 3 + 0) * 1024 + h * 64;
  const float* k = q + 1024;
  const float* v = q + 2048;
  const float* c = cosb + (size_t)n * 64;
  const float* s = sinb + (size_t)n * 64;
  float qr[64], kr[64];
  float sq = 0.f, sk = 0.f;
#pragma unroll
  for (int i = 0; i < 64; ++i) {
    int j = (i < 32) ? i + 32 : i - 32;
    float rq = (i < 32) ? -q[j] : q[j];
    float rk = (i < 32) ? -k[j] : k[j];
    float qv = q[i] * c[i] + rq * s[i];
    float kv = k[i] * c[i] + rk * s[i];
    qr[i] = qv; kr[i] = kv; sq += qv * qv; sk += kv * kv;
  }
  float qs = rsqrtf(sq * (1.f / 64.f) + 1e-5f);
  float ks = rsqrtf(sk * (1.f / 64.f) + 1e-5f);
  size_t base = ((size_t)h * 4096 + n) * 64;
#pragma unroll
  for (int i = 0; i < 64; ++i) {
    Qn[base + i] = (__bf16)(qr[i] * qs);
    Kn[base + i] = (__bf16)(kr[i] * ks);
    Vt[((size_t)h * 64 + i) * 4096 + n] = (__bf16)v[i];
  }
}

// ---------------- Streaming linear attention; writes m_concat bf16 [N][1024] ----------------
// grid (32, 16): (query tile of 128, head). 8 waves: 4(M rows) x 2(cols).
// Row-sums accumulate in per-lane registers; one cross-lane reduction at the end.
__global__ __launch_bounds__(256) void AMK_attn(const __bf16* __restrict__ Qn,
                                                const __bf16* __restrict__ Kn,
                                                const __bf16* __restrict__ Vt,
                                                const float* __restrict__ qkv,
                                                __bf16* __restrict__ mOut) {
  const int Nn = 4096;
  const int h = blockIdx.y, n0 = blockIdx.x * 128;
  __shared__ __bf16 Qs[128][72];
  __shared__ __bf16 Ks[64][72];
  __shared__ __bf16 Ws[128][72];
  __shared__ __bf16 Vs[64][72];
  __shared__ float rowsum[128];

  const int t = threadIdx.x, lane = t & 31, wave = t >> 5;
  const int wm = (wave & 3) * 32;
  const int wn = (wave >> 2) * 32;
  const int laneN = lane & 15, mOff = (lane >> 4) * 8;

  // resident Q tile 128x64
  const __bf16* Qg = Qn + ((size_t)h * Nn + n0) * 64;
#pragma unroll
  for (int i = 0; i < 4; ++i) {
    int idx = t + i * 256;
    int row = idx >> 3, col = (idx & 7) * 8;
    *(v8bf*)&Qs[row][col] = *(const v8bf*)(Qg + (size_t)row * 64 + col);
  }
  if (t < 128) rowsum[t] = 0.f;

  v8f accA[2][2];
#pragma unroll
  for (int i = 0; i < 2; ++i)
#pragma unroll
    for (int j = 0; j < 2; ++j) accA[i][j] = (v8f){0.f,0.f,0.f,0.f,0.f,0.f,0.f,0.f};
  float rs[2][8];
#pragma unroll
  for (int i = 0; i < 2; ++i)
#pragma unroll
    for (int r = 0; r < 8; ++r) rs[i][r] = 0.f;

  for (int m0 = 0; m0 < Nn; m0 += 64) {
    __syncthreads();
#pragma unroll
    for (int i = 0; i < 2; ++i) {
      int idx = t + i * 256;                 // 512 vec8 covers 64x64
      int row = idx >> 3, col = (idx & 7) * 8;
      *(v8bf*)&Ks[row][col] = *(const v8bf*)(Kn + ((size_t)h * Nn + m0 + row) * 64 + col);
      *(v8bf*)&Vs[row][col] = *(const v8bf*)(Vt + ((size_t)h * 64 + row) * Nn + m0 + col);
    }
    __syncthreads();

    // S tile (wave's 32x32): k over d=64
    v8f accS[2][2];
#pragma unroll
    for (int i = 0; i < 2; ++i)
#pragma unroll
      for (int j = 0; j < 2; ++j) accS[i][j] = (v8f){0.f,0.f,0.f,0.f,0.f,0.f,0.f,0.f};
#pragma unroll
    for (int kk = 0; kk < 64; kk += 32) {
      v16bf af[2], bfr[2];
#pragma unroll
      for (int mi = 0; mi < 2; ++mi) af[mi]  = frag_a(&Qs[0][0], 72, wm + mi * 16, kk, lane);
#pragma unroll
      for (int ni = 0; ni < 2; ++ni) bfr[ni] = frag_b(&Ks[0][0], 72, wn + ni * 16, kk, lane);
#pragma unroll
      for (int mi = 0; mi < 2; ++mi)
#pragma unroll
        for (int ni = 0; ni < 2; ++ni) accS[mi][ni] = WMMA_BF16(af[mi], bfr[ni], accS[mi][ni]);
    }

    // w = (elu(s/8)+1)^2 ; per-lane rowsum accumulation ; stash bf16 into Ws
#pragma unroll
    for (int mi = 0; mi < 2; ++mi)
#pragma unroll
      for (int ni = 0; ni < 2; ++ni)
#pragma unroll
        for (int r = 0; r < 8; ++r) {
          float sc = accS[mi][ni][r] * 0.125f;
          float w = (sc > 0.f) ? (sc + 1.f) : __expf(sc);
          w *= w;
          rs[mi][r] += w;
          Ws[wm + mi * 16 + mOff + r][wn + ni * 16 + laneN] = (__bf16)w;
        }
    __syncthreads();

    // Attr += W(128x64) * V : A from Ws rows n (k=m), B from Vs rows d (k=m)
#pragma unroll
    for (int kk = 0; kk < 64; kk += 32) {
      v16bf af[2], bfr[2];
#pragma unroll
      for (int mi = 0; mi < 2; ++mi) af[mi]  = frag_a(&Ws[0][0], 72, wm + mi * 16, kk, lane);
#pragma unroll
      for (int ni = 0; ni < 2; ++ni) bfr[ni] = frag_b(&Vs[0][0], 72, wn + ni * 16, kk, lane);
#pragma unroll
      for (int mi = 0; mi < 2; ++mi)
#pragma unroll
        for (int ni = 0; ni < 2; ++ni) accA[mi][ni] = WMMA_BF16(af[mi], bfr[ni], accA[mi][ni]);
    }
  }

  // final cross-lane rowsum reduction: 16 columns per lane group, both wn groups add
#pragma unroll
  for (int mi = 0; mi < 2; ++mi)
#pragma unroll
    for (int r = 0; r < 8; ++r) {
      float red = rs[mi][r];
      red += __shfl_xor(red, 1, 32);
      red += __shfl_xor(red, 2, 32);
      red += __shfl_xor(red, 4, 32);
      red += __shfl_xor(red, 8, 32);
      if ((lane & 15) == 0) atomicAdd(&rowsum[wm + mi * 16 + mOff + r], red);
    }
  __syncthreads();

  // epilogue: C = Attr/(rowsum+1e-6);  m = C - V ; write bf16 [n][h*64+d]
#pragma unroll
  for (int mi = 0; mi < 2; ++mi)
#pragma unroll
    for (int ni = 0; ni < 2; ++ni)
#pragma unroll
      for (int r = 0; r < 8; ++r) {
        int lrow = wm + mi * 16 + mOff + r;
        int n = n0 + lrow;
        int dcol = wn + ni * 16 + laneN;
        float cval = accA[mi][ni][r] / (rowsum[lrow] + 1e-6f);
        float vval = qkv[((size_t)n * 3 + 2) * 1024 + h * 64 + dcol];
        mOut[(size_t)n * 1024 + h * 64 + dcol] = (__bf16)(cval - vval);
      }
}

// ---------------- elementwise helpers ----------------
__global__ __launch_bounds__(256) void AMK_f2bf(const float* __restrict__ src,
                                                __bf16* __restrict__ dst, size_t n) {
  size_t id = (size_t)blockIdx.x * 256 + threadIdx.x;
  if (id < n) dst[id] = (__bf16)src[id];
}

// Tiled transpose: dst[c][r] = src[r][c]  (src R x C fp32, dst C x R bf16). R,C multiples of 32.
__global__ __launch_bounds__(256) void AMK_transpose_bf(const float* __restrict__ src,
                                                        __bf16* __restrict__ dst,
                                                        int R, int C) {
  __shared__ float tile[32][33];
  const int rb = blockIdx.y * 32, cb = blockIdx.x * 32;
  const int tx = threadIdx.x & 31, ty = threadIdx.x >> 5;   // 32 x 8
#pragma unroll
  for (int i = 0; i < 32; i += 8)
    tile[ty + i][tx] = src[(size_t)(rb + ty + i) * C + cb + tx];
  __syncthreads();
#pragma unroll
  for (int i = 0; i < 32; i += 8)
    dst[(size_t)(cb + ty + i) * R + rb + tx] = (__bf16)tile[tx][ty + i];
}

// x = X + Y; rmsnorm over row of D; write fp32 (+ optional bf16)
__global__ __launch_bounds__(256) void AMK_rms_add(const float* __restrict__ X,
                                                   const float* __restrict__ Y,
                                                   float* __restrict__ outF,
                                                   __bf16* __restrict__ out16,
                                                   int D) {
  __shared__ float red[8];
  const int n = blockIdx.x, t = threadIdx.x, lane = t & 31, wave = t >> 5;
  const float* x = X + (size_t)n * D;
  const float* y = Y + (size_t)n * D;
  float vals[4];
  float ss = 0.f;
#pragma unroll
  for (int i = 0; i < 4; ++i) {
    int c = t + i * 256;
    float v = x[c] + y[c];
    vals[i] = v; ss += v * v;
  }
  ss += __shfl_xor(ss, 1, 32);
  ss += __shfl_xor(ss, 2, 32);
  ss += __shfl_xor(ss, 4, 32);
  ss += __shfl_xor(ss, 8, 32);
  ss += __shfl_xor(ss, 16, 32);
  if (lane == 0) red[wave] = ss;
  __syncthreads();
  float tot = 0.f;
#pragma unroll
  for (int i = 0; i < 8; ++i) tot += red[i];
  float scale = rsqrtf(tot / (float)D + 1e-5f);
#pragma unroll
  for (int i = 0; i < 4; ++i) {
    int c = t + i * 256;
    float v = vals[i] * scale;
    outF[(size_t)n * D + c] = v;
    if (out16) out16[(size_t)n * D + c] = (__bf16)v;
  }
}

__global__ __launch_bounds__(256) void AMK_silu_gate(const float* __restrict__ GU,
                                                     float* __restrict__ Hf) {
  size_t id = (size_t)blockIdx.x * 256 + threadIdx.x;   // 4096*2816
  int n = (int)(id / 2816), c = (int)(id % 2816);
  float g = GU[(size_t)n * 5632 + c];
  float u = GU[(size_t)n * 5632 + 2816 + c];
  Hf[id] = (g / (1.f + __expf(-g))) * u;
}

__global__ __launch_bounds__(256) void AMK_dwconv(const float* __restrict__ Hf,
                                                  const float* __restrict__ w,
                                                  const float* __restrict__ b,
                                                  __bf16* __restrict__ Hc16) {
  size_t id = (size_t)blockIdx.x * 256 + threadIdx.x;   // 4096*2816
  int n = (int)(id / 2816), c = (int)(id % 2816);
  float acc = b[c];
#pragma unroll
  for (int j = 0; j < 3; ++j) {
    int nn = n - 1 + j;
    if (nn >= 0 && nn < 4096) acc += Hf[(size_t)nn * 2816 + c] * w[j * 2816 + c];
  }
  Hc16[id] = (__bf16)(acc / (1.f + __expf(-acc)));
}

// ---------------- host orchestration ----------------
extern "C" void kernel_launch(void* const* d_in, const int* in_sizes, int n_in,
                              void* d_out, int out_size, void* d_ws, size_t ws_size,
                              hipStream_t stream) {
  (void)in_sizes; (void)n_in; (void)out_size; (void)ws_size;
  const float* Q_in   = (const float*)d_in[0];
  const float* cosb   = (const float*)d_in[1];
  const float* sinb   = (const float*)d_in[2];
  const float* W_qkv  = (const float*)d_in[3];
  const float* W_o    = (const float*)d_in[4];
  const float* W_up   = (const float*)d_in[5];
  const float* conv_w = (const float*)d_in[6];
  const float* conv_b = (const float*)d_in[7];
  const float* W_down = (const float*)d_in[8];
  float* out = (float*)d_out;

  const int Nn = 4096, D = 1024, INNER = 2816, NUP = 5632, NQKV = 3072;

  char* base = (char*)d_ws;
  size_t off = 0;
  auto alloc = [&](size_t bytes) -> void* {
    void* p = base + off;
    off = (off + bytes + 255) & ~(size_t)255;
    return p;
  };

  float*  bigF    = (float*)alloc((size_t)Nn * NUP * 4);        // qkv fp32, later GU fp32
  __bf16* X16     = (__bf16*)alloc((size_t)Nn * D * 2);
  __bf16* Wqkv_t  = (__bf16*)alloc((size_t)NQKV * D * 2);
  __bf16* Wo_t    = (__bf16*)alloc((size_t)D * D * 2);
  __bf16* Wup_t   = (__bf16*)alloc((size_t)NUP * D * 2);
  __bf16* Wdown_t = (__bf16*)alloc((size_t)D * INNER * 2);
  __bf16* Qn      = (__bf16*)alloc((size_t)16 * Nn * 64 * 2);
  __bf16* Kn      = (__bf16*)alloc((size_t)16 * Nn * 64 * 2);
  __bf16* Vt      = (__bf16*)alloc((size_t)16 * 64 * Nn * 2);
  __bf16* m16     = (__bf16*)alloc((size_t)Nn * D * 2);
  float*  OY      = (float*)alloc((size_t)Nn * D * 4);          // W_o out, later W_down out
  float*  QiF     = (float*)alloc((size_t)Nn * D * 4);
  __bf16* Qi16    = (__bf16*)alloc((size_t)Nn * D * 2);
  float*  Hf      = (float*)alloc((size_t)Nn * INNER * 4);
  __bf16* Hc16    = (__bf16*)alloc((size_t)Nn * INNER * 2);

  // conversions (coalesced tiled transposes)
  AMK_f2bf<<<(Nn * D) / 256, 256, 0, stream>>>(Q_in, X16, (size_t)Nn * D);
  AMK_transpose_bf<<<dim3(NQKV / 32, D / 32), 256, 0, stream>>>(W_qkv, Wqkv_t, D, NQKV);
  AMK_transpose_bf<<<dim3(D / 32, D / 32), 256, 0, stream>>>(W_o, Wo_t, D, D);
  AMK_transpose_bf<<<dim3(NUP / 32, D / 32), 256, 0, stream>>>(W_up, Wup_t, D, NUP);
  AMK_transpose_bf<<<dim3(D / 32, INNER / 32), 256, 0, stream>>>(W_down, Wdown_t, INNER, D);

  // qkv = X @ W_qkv
  AMK_gemm_bf16<<<dim3(NQKV / 256, Nn / 128), 256, 0, stream>>>(X16, Wqkv_t, bigF, Nn, NQKV, D);
  // RoPE + per-head RMS + V transpose
  AMK_rope_norm<<<(Nn * 16) / 256, 256, 0, stream>>>(bigF, cosb, sinb, Qn, Kn, Vt);
  // streaming attention -> m_concat bf16
  AMK_attn<<<dim3(Nn / 128, 16), 256, 0, stream>>>(Qn, Kn, Vt, bigF, m16);
  // O = m @ W_o ; Q_interact = rms(Q_in + O)
  AMK_gemm_bf16<<<dim3(D / 256, Nn / 128), 256, 0, stream>>>(m16, Wo_t, OY, Nn, D, D);
  AMK_rms_add<<<Nn, 256, 0, stream>>>(Q_in, OY, QiF, Qi16, D);
  // GU = Qi @ W_up ; Hf = silu(G)*U
  AMK_gemm_bf16<<<dim3(NUP / 256, Nn / 128), 256, 0, stream>>>(Qi16, Wup_t, bigF, Nn, NUP, D);
  AMK_silu_gate<<<(Nn * INNER) / 256, 256, 0, stream>>>(bigF, Hf);
  // depthwise conv + silu -> bf16
  AMK_dwconv<<<(Nn * INNER) / 256, 256, 0, stream>>>(Hf, conv_w, conv_b, Hc16);
  // Y = Hc @ W_down ; out = rms(Qi + Y)
  AMK_gemm_bf16<<<dim3(D / 256, Nn / 128), 256, 0, stream>>>(Hc16, Wdown_t, OY, Nn, D, INNER);
  AMK_rms_add<<<Nn, 256, 0, stream>>>(QiF, OY, out, nullptr, D);
}
